// CausalAttention_71691594105150
// MI455X (gfx1250) — compile-verified, matrix operand order checked by reading, not run on hardware
//
#include <hip/hip_runtime.h>

// Causal attention, B=4, S=4096, d=1024, fp32 I/O.
// fp32 -> bf16 once (W transposed), WMMA bf16 QKV GEMMs (16x64 per wave,
// software-pipelined k-loop), flash attention with TDM (tensor_load_to_lds)
// staging of Q/K tiles and online softmax in f32.
// ~240 GF vs ~250 MB HBM -> compute-bound -> bf16 WMMA everywhere.

#define DMODEL 1024
#define SEQ    4096
#define NROWS  16384

typedef __attribute__((ext_vector_type(16))) __bf16 v16bf;
typedef __attribute__((ext_vector_type(8)))  __bf16 v8bf;
typedef __attribute__((ext_vector_type(8)))  float  v8f;
typedef __attribute__((ext_vector_type(4)))  unsigned int u32x4;
typedef __attribute__((ext_vector_type(8)))  int i32x8;
typedef __attribute__((ext_vector_type(4)))  int i32x4;

#if defined(__gfx1250__) && __has_builtin(__builtin_amdgcn_tensor_load_to_lds) && \
    __has_builtin(__builtin_amdgcn_s_wait_tensorcnt)
#define HAVE_TDM 1
#endif

#ifdef HAVE_TDM
// 1-D contiguous DMA: global -> LDS, nelem2B bf16 elements (< 65536).
// D# per CDNA5 ISA (groups zero-padded; 1-D tile).
__device__ __forceinline__ void tdm_load_1d(void* lds_dst, const void* gsrc,
                                            unsigned nelem2B) {
  unsigned long long ga = (unsigned long long)gsrc;
  // generic LDS pointer: addr[31:0] is the LDS byte address (aperture rule)
  unsigned lds_off = (unsigned)(unsigned long long)lds_dst;
  u32x4 g0;
  g0[0] = 1u;                                             // count=1 (valid D#)
  g0[1] = lds_off;                                        // lds_addr
  g0[2] = (unsigned)ga;                                   // global_addr[31:0]
  g0[3] = (unsigned)((ga >> 32) & 0x01FFFFFFu) | (2u << 30); // [56:32] | type=2
  i32x8 g1;
  g1[0] = 0x00010000;                        // data_size=1 -> 2 bytes, mask=0
  g1[1] = (int)((nelem2B & 0xFFFFu) << 16);  // tensor_dim0[15:0]
  g1[2] = (int)(nelem2B >> 16);              // tensor_dim0[31:16]
  g1[3] = (int)((nelem2B & 0xFFFFu) << 16);  // tile_dim0
  g1[4] = 0; g1[5] = 0; g1[6] = 0; g1[7] = 0;
  i32x4 z4 = {0, 0, 0, 0};
#if __clang_major__ >= 23
  i32x8 z8 = {0, 0, 0, 0, 0, 0, 0, 0};
  __builtin_amdgcn_tensor_load_to_lds(g0, g1, z4, z4, z8, 0);
#else
  __builtin_amdgcn_tensor_load_to_lds(g0, g1, z4, z4, 0);
#endif
}
#endif

// WMMA operand from two contiguous 16-byte chunks (lane L<16: K {0..7,16..23},
// lane L>=16: K {8..15,24..31} of its row/col -> each half is 16 B contiguous).
__device__ __forceinline__ v16bf ld_pair(const __bf16* p0, const __bf16* p1) {
  v8bf lo = *(const v8bf*)p0;
  v8bf hi = *(const v8bf*)p1;
  v16bf r;
#pragma unroll
  for (int i = 0; i < 8; ++i) { r[i] = lo[i]; r[i + 8] = hi[i]; }
  return r;
}

#define WMMA_BF16(A, B, C, RA) \
  __builtin_amdgcn_wmma_f32_16x16x32_bf16(false, (A), false, (B), (short)0, (C), (RA), false)

__global__ void cvt_f32_bf16(const float* __restrict__ in,
                             __bf16* __restrict__ out, int n) {
  int i = blockIdx.x * blockDim.x + threadIdx.x;
  if (i < n) out[i] = (__bf16)in[i];
}

// out[c][r] = (bf16) in[r][c], 1024x1024, 32x32 LDS tiles (both sides coalesced)
__global__ void cvt_transpose_bf16(const float* __restrict__ in,
                                   __bf16* __restrict__ out) {
  __shared__ float tile[32][33];
  const int tx = threadIdx.x & 31, ty = threadIdx.x >> 5;  // 32 x 8
  const int bx = blockIdx.x, by = blockIdx.y;
#pragma unroll
  for (int r = 0; r < 32; r += 8)
    tile[ty + r][tx] = in[(size_t)(by * 32 + ty + r) * DMODEL + bx * 32 + tx];
  __syncthreads();
#pragma unroll
  for (int r = 0; r < 32; r += 8)
    out[(size_t)(bx * 32 + ty + r) * DMODEL + by * 32 + tx] =
        (__bf16)tile[tx][ty + r];
}

// Y[N x 1024] = X[N x 1024] @ W, with Wt = W^T (bf16, [d_out][d_in]).
// 8 waves/block, each wave a 16x64 tile.  One-deep software pipeline: the
// next k-step's 10 global_load_b128 are issued before the current 4 WMMAs,
// hiding load latency behind the matrix pipe (no full loadcnt drains).
__global__ void gemm_bf16(const __bf16* __restrict__ X,
                          const __bf16* __restrict__ Wt,
                          __bf16* __restrict__ Y) {
  const int lane = threadIdx.x & 31;
  const int wv   = threadIdx.x >> 5;
  const int tile = blockIdx.x * 8 + wv;   // 16384 wave-tiles
  const int mtile = tile >> 4;            // 1024 M-tiles
  const int nmac  = tile & 15;            // 16 macro-tiles of 64 cols
  const int l16  = lane & 15;
  const int half = lane >> 4;
  const int ks0  = half * 8;

  const __bf16* xrow = X + (size_t)(mtile * 16 + l16) * DMODEL + ks0;
  const __bf16* w0   = Wt + (size_t)(nmac * 64 + l16) * DMODEL + ks0;

  v8f acc0 = {}, acc1 = {}, acc2 = {}, acc3 = {};

  // prologue: operands for kb = 0
  v16bf a  = ld_pair(xrow, xrow + 16);
  v16bf b0 = ld_pair(w0 +  0 * DMODEL, w0 +  0 * DMODEL + 16);
  v16bf b1 = ld_pair(w0 + 16 * DMODEL, w0 + 16 * DMODEL + 16);
  v16bf b2 = ld_pair(w0 + 32 * DMODEL, w0 + 32 * DMODEL + 16);
  v16bf b3 = ld_pair(w0 + 48 * DMODEL, w0 + 48 * DMODEL + 16);

#pragma unroll 2
  for (int kb = 0; kb < DMODEL - 32; kb += 32) {
    const int kn = kb + 32;
    // prefetch next step's operands (independent of current WMMAs)
    v16bf an  = ld_pair(xrow + kn, xrow + kn + 16);
    v16bf b0n = ld_pair(w0 +  0 * DMODEL + kn, w0 +  0 * DMODEL + kn + 16);
    v16bf b1n = ld_pair(w0 + 16 * DMODEL + kn, w0 + 16 * DMODEL + kn + 16);
    v16bf b2n = ld_pair(w0 + 32 * DMODEL + kn, w0 + 32 * DMODEL + kn + 16);
    v16bf b3n = ld_pair(w0 + 48 * DMODEL + kn, w0 + 48 * DMODEL + kn + 16);
    acc0 = WMMA_BF16(a, b0, acc0, false);
    acc1 = WMMA_BF16(a, b1, acc1, true);
    acc2 = WMMA_BF16(a, b2, acc2, true);
    acc3 = WMMA_BF16(a, b3, acc3, true);
    a = an; b0 = b0n; b1 = b1n; b2 = b2n; b3 = b3n;
  }
  // epilogue: last k-step
  acc0 = WMMA_BF16(a, b0, acc0, false);
  acc1 = WMMA_BF16(a, b1, acc1, true);
  acc2 = WMMA_BF16(a, b2, acc2, true);
  acc3 = WMMA_BF16(a, b3, acc3, true);

  const int orow = mtile * 16 + half * 8;
  v8f* accs[4] = {&acc0, &acc1, &acc2, &acc3};
#pragma unroll
  for (int t = 0; t < 4; ++t) {
    const int col = nmac * 64 + t * 16 + l16;
#pragma unroll
    for (int j = 0; j < 8; ++j)
      Y[(size_t)(orow + j) * DMODEL + col] = (__bf16)((*accs[t])[j]);
  }
}

// Flash attention: 1 block (8 waves) per 16-query tile; wave owns 128 out dims.
__global__ void flash_attn(const __bf16* __restrict__ Q,
                           const __bf16* __restrict__ K,
                           const __bf16* __restrict__ V,
                           float* __restrict__ Out) {
  extern __shared__ char smem[];
  __bf16* sQ  = (__bf16*)smem;               // [16][1024]  32 KB
  __bf16* sK  = sQ + 16 * 1024;              // [32][1024]  64 KB
  __bf16* sVt = sK + 32 * 1024;              // [1024][32]  64 KB (transposed)
  float*  sS  = (float*)(sVt + 32 * 1024);   // [16][32]    2 KB

  const int tid  = threadIdx.x;
  const int lane = tid & 31;
  const int wv   = tid >> 5;
  const int l16  = lane & 15;
  const int half = lane >> 4;
  const int ks0  = half * 8;

  const int qt    = blockIdx.x;
  const int qrow0 = qt * 16;
  const int batch = qrow0 >> 12;
  const int q0    = qrow0 & 4095;
  const size_t bbase = (size_t)batch * SEQ * DMODEL;

  // Q tile -> LDS (TDM if available, overlapped with first K tile DMA)
#ifdef HAVE_TDM
  if (wv == 0) tdm_load_1d(sQ, Q + (size_t)qrow0 * DMODEL, 16 * 1024);
#else
  {
    const uint4* src = (const uint4*)(Q + (size_t)qrow0 * DMODEL);
    uint4* dst = (uint4*)sQ;
    for (int i = tid; i < 2048; i += 256) dst[i] = src[i];
  }
#endif

  v8f acc[8] = {};
  float m_run = -INFINITY, l_run = 0.0f;
  const float scale = 0.03125f;   // 1/sqrt(1024)
  const int dbase = wv * 128;

  for (int kb = 0; kb < q0 + 16; kb += 32) {
    __syncthreads();
    // K tile: 64 KB contiguous (TDM DMA overlaps the manual V transpose below)
#ifdef HAVE_TDM
    if (wv == 0) tdm_load_1d(sK, K + bbase + (size_t)kb * DMODEL, 32 * 1024);
#else
    {
      const uint4* src = (const uint4*)(K + bbase + (size_t)kb * DMODEL);
      uint4* dst = (uint4*)sK;
      for (int i = tid; i < 4096; i += 256) dst[i] = src[i];
    }
#endif
    // V tile, transposed (PV B-operand becomes two contiguous 16B DS loads)
    {
      const __bf16* src = V + bbase + (size_t)kb * DMODEL;
      for (int i = tid; i < 32 * 1024; i += 256) {
        int kr = i >> 10, d = i & 1023;
        sVt[d * 32 + kr] = src[i];
      }
    }
    for (int i = tid; i < 16 * 32; i += 256) sS[i] = 0.0f;
#ifdef HAVE_TDM
    if (wv == 0) __builtin_amdgcn_s_wait_tensorcnt(0);
#endif
    __syncthreads();

    // Partial scores over this wave's 128 dims; A (Q) reused across the two
    // key sub-tiles (reuse_a hint).
    v8f ps0 = {}, ps1 = {};
#pragma unroll
    for (int dc = 0; dc < 4; ++dc) {
      const int db = dbase + dc * 32;
      v16bf a  = ld_pair(&sQ[l16 * 1024 + db + ks0],
                         &sQ[l16 * 1024 + db + ks0 + 16]);
      v16bf b0 = ld_pair(&sK[l16 * 1024 + db + ks0],
                         &sK[l16 * 1024 + db + ks0 + 16]);
      v16bf b1 = ld_pair(&sK[(16 + l16) * 1024 + db + ks0],
                         &sK[(16 + l16) * 1024 + db + ks0 + 16]);
      ps0 = WMMA_BF16(a, b0, ps0, false);
      ps1 = WMMA_BF16(a, b1, ps1, true);
    }
#pragma unroll
    for (int j = 0; j < 8; ++j) {   // cross-wave reduce via ds_add_f32
      atomicAdd(&sS[(half * 8 + j) * 32 + l16],      ps0[j]);
      atomicAdd(&sS[(half * 8 + j) * 32 + 16 + l16], ps1[j]);
    }
    __syncthreads();

    // Online softmax: lane pair (l16, l16+16) splits the 32 cols of row l16
    const int row  = l16;
    const int qpos = q0 + row;
    float sv[16];
    float mloc = -INFINITY;
#pragma unroll
    for (int i = 0; i < 8; ++i) {
      const int c0 = ks0 + i, c1 = 16 + ks0 + i;
      float x0 = (kb + c0 <= qpos) ? sS[row * 32 + c0] * scale : -INFINITY;
      float x1 = (kb + c1 <= qpos) ? sS[row * 32 + c1] * scale : -INFINITY;
      sv[i] = x0; sv[8 + i] = x1;
      mloc = fmaxf(mloc, fmaxf(x0, x1));
    }
    mloc = fmaxf(mloc, __shfl_xor(mloc, 16, 32));
    const float m_new = fmaxf(m_run, mloc);
    const float alpha = __expf(m_run - m_new);
    float rs = 0.0f;
    v16bf pa;   // P already in A-operand layout
#pragma unroll
    for (int i = 0; i < 16; ++i) {
      float p = __expf(sv[i] - m_new);
      rs += p;
      pa[i] = (__bf16)p;
    }
    rs += __shfl_xor(rs, 16, 32);
    l_run = l_run * alpha + rs;
    m_run = m_new;

#pragma unroll
    for (int j = 0; j < 8; ++j) {
      const float fj = __shfl(alpha, half ? (j + 8) : j, 32);
#pragma unroll
      for (int c = 0; c < 8; ++c) acc[c][j] *= fj;
    }

    // P x V: one WMMA per 16-dim chunk; A (P) reused across all 8 (reuse_a).
#define PV_STEP(c, ra)                                                        \
    {                                                                         \
      const __bf16* vb = &sVt[(size_t)(dbase + (c) * 16 + l16) * 32 + ks0];   \
      v16bf b = ld_pair(vb, vb + 16);                                         \
      acc[c] = WMMA_BF16(pa, b, acc[c], ra);                                  \
    }
    PV_STEP(0, false) PV_STEP(1, true) PV_STEP(2, true) PV_STEP(3, true)
    PV_STEP(4, true)  PV_STEP(5, true) PV_STEP(6, true) PV_STEP(7, true)
#undef PV_STEP
  }

  // Normalize and store fp32
#pragma unroll
  for (int j = 0; j < 8; ++j) {
    const float lr  = __shfl(l_run, half ? (j + 8) : j, 32);
    const float inv = 1.0f / lr;
    const int orow  = qrow0 + half * 8 + j;
#pragma unroll
    for (int c = 0; c < 8; ++c)
      Out[(size_t)orow * DMODEL + dbase + c * 16 + l16] = acc[c][j] * inv;
  }
}

extern "C" void kernel_launch(void* const* d_in, const int* in_sizes, int n_in,
                              void* d_out, int out_size, void* d_ws, size_t ws_size,
                              hipStream_t stream) {
  const float* x  = (const float*)d_in[0];
  const float* Wq = (const float*)d_in[1];
  const float* Wk = (const float*)d_in[2];
  const float* Wv = (const float*)d_in[3];
  float* out = (float*)d_out;

  // Workspace (bf16), ~141 MB: xb 32MB | WqT/WkT/WvT 2MB | Qb/Kb/Vb 32MB each
  char* ws = (char*)d_ws;
  __bf16* xb   = (__bf16*)(ws + 0);
  __bf16* wqtb = (__bf16*)(ws + 33554432);
  __bf16* wktb = (__bf16*)(ws + 35651584);
  __bf16* wvtb = (__bf16*)(ws + 37748736);
  __bf16* Qb   = (__bf16*)(ws + 39845888);
  __bf16* Kb   = (__bf16*)(ws + 73400320);
  __bf16* Vb   = (__bf16*)(ws + 106954752);

  cvt_f32_bf16<<<(NROWS * DMODEL) / 256, 256, 0, stream>>>(x, xb, NROWS * DMODEL);
  dim3 tgrid(32, 32);
  cvt_transpose_bf16<<<tgrid, 256, 0, stream>>>(Wq, wqtb);
  cvt_transpose_bf16<<<tgrid, 256, 0, stream>>>(Wk, wktb);
  cvt_transpose_bf16<<<tgrid, 256, 0, stream>>>(Wv, wvtb);

  // 16384 wave-tiles (16x64 each) / 8 waves per block = 2048 blocks per GEMM
  gemm_bf16<<<2048, 256, 0, stream>>>(xb, wqtb, Qb);
  gemm_bf16<<<2048, 256, 0, stream>>>(xb, wktb, Kb);
  gemm_bf16<<<2048, 256, 0, stream>>>(xb, wvtb, Vb);

  // 1024 query tiles; 166 KB dynamic LDS per workgroup (<= 320 KB/WGP)
  flash_attn<<<1024, 256, 165888, stream>>>(Qb, Kb, Vb, out);
}
